// Nanofold_7026566497082
// MI455X (gfx1250) — matrix operand order repaired and checked by reading, not compiled, not run on hardware
//
#include <hip/hip_runtime.h>
#include <hip/hip_bf16.h>

#define N 384
#define D 128
#define H 2
#define DH 64
#define M (N * N)                    // 147456 flattened rows
#define SCALE 0.08838834764831845f   // D^-0.5 (torch scales by full dim)
#define SS 388                       // padded S row stride (f32)
#define VS 408                       // padded V^T row stride (bf16), 16B-aligned

typedef __attribute__((ext_vector_type(16))) __bf16 v16bf;
typedef __attribute__((ext_vector_type(8)))  __bf16 v8bf;
typedef __attribute__((ext_vector_type(4)))  __bf16 v4bf;
typedef __attribute__((ext_vector_type(8)))  float  v8f;

static __device__ __forceinline__ v16bf cat16(v8bf lo, v8bf hi) {
    return __builtin_shufflevector(lo, hi, 0, 1, 2, 3, 4, 5, 6, 7,
                                   8, 9, 10, 11, 12, 13, 14, 15);
}

// ---------------------------------------------------------------------------
// Kernel 0: convert 5 weight matrices to bf16 AND swizzle into WMMA B-fragment
// order: [matrix][kc(4)][coltile(8)][lane(32)][e(16)] contiguous per lane.
// ---------------------------------------------------------------------------
__global__ void cvt_weights(const float* Wq, const float* Wk, const float* Wv,
                            const float* Wg, const float* Wo, __bf16* o) {
    int i = blockIdx.x * 256 + threadIdx.x;          // 0 .. 5*16384-1
    if (i >= 5 * D * D) return;
    int m = i >> 14, j = i & 16383;
    int e    = j & 15;
    int lane = (j >> 4) & 31;
    int ct   = (j >> 9) & 7;
    int kc   = j >> 12;
    int lh = lane >> 4, l16 = lane & 15;
    int k    = kc * 32 + (e & 8) * 2 + lh * 8 + (e & 7);
    int ncol = ct * 16 + l16;
    const float* src = (m == 0) ? Wq : (m == 1) ? Wk : (m == 2) ? Wv
                     : (m == 3) ? Wg : Wo;
    o[i] = (__bf16)src[k * D + ncol];
}

// ---------------------------------------------------------------------------
// Kernel 1: fused Q/K/V/gate projection + bias GEMV (bias stored as [h][m]).
// 64-row blocks: weight B-fragments loaded once, reused over 4 row subtiles.
// ---------------------------------------------------------------------------
__global__ __launch_bounds__(256) void proj_kernel(
    const float* __restrict__ pair, const float* __restrict__ Wb,
    const __bf16* __restrict__ wts,
    __bf16* __restrict__ qw, __bf16* __restrict__ kw,
    __bf16* __restrict__ vw, __bf16* __restrict__ gw,
    float* __restrict__ biasw) {
    __shared__ __bf16 sA[64 * D];                    // 16 KB
    const int tid = threadIdx.x;
    const size_t rowbase = (size_t)blockIdx.x * 64;

    // stage 64x128 input tile: float4 loads -> v4bf stores
#pragma unroll
    for (int it = 0; it < 8; ++it) {
        int g = tid + 256 * it;                      // group of 4 elems
        float4 v = *(const float4*)&pair[rowbase * D + g * 4];
        v4bf c;
        c[0] = (__bf16)v.x; c[1] = (__bf16)v.y;
        c[2] = (__bf16)v.z; c[3] = (__bf16)v.w;
        *(v4bf*)&sA[g * 4] = c;
    }
    __syncthreads();

    const int wave = tid >> 5, lane = tid & 31;
    const int l16 = lane & 15, lhalf = lane >> 4;
    const int col = wave * 16 + l16;

#pragma unroll
    for (int m = 0; m < 4; ++m) {                    // q, k, v, gate
        const __bf16* W = wts + m * D * D;
        v16bf bfrag[4];
#pragma unroll
        for (int kc = 0; kc < 4; ++kc)               // load once, reuse 4x
            bfrag[kc] = *(const v16bf*)(W + ((kc * 8 + wave) * 32 + lane) * 16);
        __bf16* dst = (m == 0) ? qw : (m == 1) ? kw : (m == 2) ? vw : gw;
#pragma unroll
        for (int rt = 0; rt < 4; ++rt) {             // 4 row subtiles
            const __bf16* arow = &sA[(rt * 16 + l16) * D];
            v8f acc = {};
#pragma unroll
            for (int kc = 0; kc < 4; ++kc) {
                v8bf lo = *(const v8bf*)(arow + kc * 32 + lhalf * 8);
                v8bf hi = *(const v8bf*)(arow + kc * 32 + 16 + lhalf * 8);
                acc = __builtin_amdgcn_wmma_f32_16x16x32_bf16(
                    false, cat16(lo, hi), false, bfrag[kc], (short)0, acc,
                    false, false);
            }
#pragma unroll
            for (int r = 0; r < 8; ++r) {
                int row = rt * 16 + r + lhalf * 8;
                float x = acc[r];
                if (m == 3) x = 1.0f / (1.0f + __expf(-x));   // gate sigmoid
                dst[(rowbase + row) * D + col] = (__bf16)x;
            }
        }
    }

    // bias = pair @ Wb  (stored transposed: biasw[h*M + m]); waves 0-3
    if (wave < 4) {
        int row = wave * 16 + l16;
        float s = 0.f;
        for (int kk = 0; kk < D; ++kk)
            s += (float)sA[row * D + kk] * Wb[kk * H + lhalf];
        biasw[(size_t)lhalf * M + rowbase + row] = s;
    }
}

// ---------------------------------------------------------------------------
// Kernel 2: attention for one (qb, h, n).
// ---------------------------------------------------------------------------
__global__ __launch_bounds__(256) void attn_kernel(
    const __bf16* __restrict__ qw, const __bf16* __restrict__ kw,
    const __bf16* __restrict__ vw, const float* __restrict__ biasw,
    __bf16* __restrict__ aw) {
    const int qb = blockIdx.x;      // 0..5  (64-row q block)
    const int h  = blockIdx.y;      // 0..1
    const int n  = blockIdx.z;      // 0..383

    __shared__ float  S[64 * SS];   // score/prob tile (~97 KB)
    __shared__ float  red[64 * 8];  // softmax partials
    __shared__ __bf16 sVt[DH * VS]; // V transposed: [d][kr] (~51 KB)

    const int tid = threadIdx.x;
    const int wave = tid >> 5, lane = tid & 31;
    const int l16 = lane & 15, lhalf = lane >> 4;
    const int qs = wave & 3;

    const __bf16* Qp = qw + (size_t)n * N * D + h * DH;
    const __bf16* Kp = kw + (size_t)n * N * D + h * DH;
    const __bf16* Vp = vw + (size_t)n * N * D + h * DH;

    // ---- Prologue: bias -> S (coalesced float4), V -> sVt (transposed) ----
    {
        const float* bsrc = biasw + (size_t)h * M + (size_t)qb * 64 * N;
#pragma unroll
        for (int it = 0; it < 24; ++it) {
            int g = tid + 256 * it;                 // float4 group, 6144 total
            int r = g / (N / 4), c4 = (g % (N / 4)) * 4;
            *(float4*)&S[r * SS + c4] = *(const float4*)&bsrc[g * 4];
        }
#pragma unroll
        for (int it = 0; it < 6; ++it) {
            int g = tid + 256 * it;                 // 16-elem group, 1536 total
            int kr = g >> 2, d0 = (g & 3) * 16;
            v16bf v = *(const v16bf*)(Vp + (size_t)kr * D + d0);
#pragma unroll
            for (int e = 0; e < 16; ++e)
                sVt[(d0 + e) * VS + kr] = v[e];
        }
    }

    // All 8 Q fragments (4 q-subtiles x 2 K32 chunks) in registers
    v16bf aqf[4][2];
#pragma unroll
    for (int q4 = 0; q4 < 4; ++q4) {
        const __bf16* qr = Qp + (size_t)(qb * 64 + q4 * 16 + l16) * D;
#pragma unroll
        for (int kc = 0; kc < 2; ++kc) {
            v8bf lo = *(const v8bf*)(qr + kc * 32 + lhalf * 8);
            v8bf hi = *(const v8bf*)(qr + kc * 32 + 16 + lhalf * 8);
            aqf[q4][kc] = cat16(lo, hi);
        }
    }
    __syncthreads();

    // ---- Phase 1: S = QK^T * scale + bias.  Each wave owns 3 k-tiles and
    //      computes all 4 q-subtiles against them (K loaded once per block).
    for (int t = 0; t < 3; ++t) {
        const int kt = wave + 8 * t;                 // 24 k-tiles
        const int ki = kt * 16 + l16;
        const __bf16* krow = Kp + (size_t)ki * D;
        v16bf bk[2];
#pragma unroll
        for (int kc = 0; kc < 2; ++kc) {
            v8bf lo = *(const v8bf*)(krow + kc * 32 + lhalf * 8);
            v8bf hi = *(const v8bf*)(krow + kc * 32 + 16 + lhalf * 8);
            bk[kc] = cat16(lo, hi);
        }
        v8f acc[4] = {};
#pragma unroll
        for (int q4 = 0; q4 < 4; ++q4)
#pragma unroll
            for (int kc = 0; kc < 2; ++kc)
                acc[q4] = __builtin_amdgcn_wmma_f32_16x16x32_bf16(
                    false, aqf[q4][kc], false, bk[kc], (short)0, acc[q4],
                    false, false);
#pragma unroll
        for (int q4 = 0; q4 < 4; ++q4)
#pragma unroll
            for (int r = 0; r < 8; ++r) {
                int idx = (q4 * 16 + r + lhalf * 8) * SS + ki;
                S[idx] = acc[q4][r] * SCALE + S[idx];
            }
    }
    __syncthreads();

    // ---- Phase 2: softmax over k (4 threads per row, float4 vectorized) ----
    {
        const int r = tid & 63, qtr = tid >> 6;
        float* row = &S[r * SS + qtr * 96];
        float4 m4 = {-3.0e38f, -3.0e38f, -3.0e38f, -3.0e38f};
#pragma unroll
        for (int i = 0; i < 24; ++i) {
            float4 v = *(const float4*)(row + i * 4);
            m4.x = fmaxf(m4.x, v.x); m4.y = fmaxf(m4.y, v.y);
            m4.z = fmaxf(m4.z, v.z); m4.w = fmaxf(m4.w, v.w);
        }
        red[r * 8 + qtr] = fmaxf(fmaxf(m4.x, m4.y), fmaxf(m4.z, m4.w));
        __syncthreads();
        float mx = fmaxf(fmaxf(red[r * 8 + 0], red[r * 8 + 1]),
                         fmaxf(red[r * 8 + 2], red[r * 8 + 3]));
        float sum = 0.f;
#pragma unroll
        for (int i = 0; i < 24; ++i) {
            float4 v = *(const float4*)(row + i * 4);
            v.x = __expf(v.x - mx); v.y = __expf(v.y - mx);
            v.z = __expf(v.z - mx); v.w = __expf(v.w - mx);
            *(float4*)(row + i * 4) = v;
            sum += v.x + v.y + v.z + v.w;
        }
        red[r * 8 + 4 + qtr] = sum;
        __syncthreads();
        sum = red[r * 8 + 4] + red[r * 8 + 5] + red[r * 8 + 6] + red[r * 8 + 7];
        float inv = 1.0f / sum;
#pragma unroll
        for (int i = 0; i < 24; ++i) {
            float4 v = *(const float4*)(row + i * 4);
            v.x *= inv; v.y *= inv; v.z *= inv; v.w *= inv;
            *(float4*)(row + i * 4) = v;
        }
    }
    __syncthreads();

    // ---- Phase 3: O = P @ V ----
    const int dtb = (wave >> 2) * 2;
#pragma unroll
    for (int u = 0; u < 2; ++u) {
        const int dt = dtb + u;                       // dh tile (0..3)
        v8f acc = {};
        for (int kc = 0; kc < 12; ++kc) {
            // A = P fragment: 4 contiguous float4 LDS loads + cvt
            const float* prow = &S[(qs * 16 + l16) * SS + kc * 32];
            float4 p0 = *(const float4*)(prow + lhalf * 8);
            float4 p1 = *(const float4*)(prow + lhalf * 8 + 4);
            float4 p2 = *(const float4*)(prow + 16 + lhalf * 8);
            float4 p3 = *(const float4*)(prow + 16 + lhalf * 8 + 4);
            v16bf a;
            a[0]  = (__bf16)p0.x; a[1]  = (__bf16)p0.y;
            a[2]  = (__bf16)p0.z; a[3]  = (__bf16)p0.w;
            a[4]  = (__bf16)p1.x; a[5]  = (__bf16)p1.y;
            a[6]  = (__bf16)p1.z; a[7]  = (__bf16)p1.w;
            a[8]  = (__bf16)p2.x; a[9]  = (__bf16)p2.y;
            a[10] = (__bf16)p2.z; a[11] = (__bf16)p2.w;
            a[12] = (__bf16)p3.x; a[13] = (__bf16)p3.y;
            a[14] = (__bf16)p3.z; a[15] = (__bf16)p3.w;
            // B = V fragment: contiguous v8bf LDS loads from transposed V
            const __bf16* vrow = &sVt[(dt * 16 + l16) * VS + kc * 32 + lhalf * 8];
            v8bf lo = *(const v8bf*)(vrow);
            v8bf hi = *(const v8bf*)(vrow + 16);
            acc = __builtin_amdgcn_wmma_f32_16x16x32_bf16(
                false, a, false, cat16(lo, hi), (short)0, acc, false, false);
        }
#pragma unroll
        for (int r = 0; r < 8; ++r) {
            int qg = qb * 64 + qs * 16 + r + lhalf * 8;
            int dcol = dt * 16 + l16;
            aw[(size_t)(n * N + qg) * D + h * DH + dcol] = (__bf16)acc[r];
        }
    }
}

// ---------------------------------------------------------------------------
// Kernel 3: out = (attn @ Wo) * gate  -> f32 output.  64-row blocks,
// Wo fragments loaded once and reused over 4 row subtiles.
// ---------------------------------------------------------------------------
__global__ __launch_bounds__(256) void outproj_kernel(
    const __bf16* __restrict__ aw, const __bf16* __restrict__ wts,
    const __bf16* __restrict__ gw, float* __restrict__ out) {
    __shared__ __bf16 sA[64 * D];                    // 16 KB
    const int tid = threadIdx.x;
    const size_t rowbase = (size_t)blockIdx.x * 64;
#pragma unroll
    for (int it = 0; it < 4; ++it) {
        int g = tid + 256 * it;                      // v8bf group, 1024 total
        *(v8bf*)&sA[g * 8] = *(const v8bf*)&aw[rowbase * D + g * 8];
    }
    __syncthreads();

    const int wave = tid >> 5, lane = tid & 31;
    const int l16 = lane & 15, lhalf = lane >> 4;
    const __bf16* W = wts + 4 * D * D;               // swizzled Wo
    const int col = wave * 16 + l16;

    v16bf bfrag[4];
#pragma unroll
    for (int kc = 0; kc < 4; ++kc)
        bfrag[kc] = *(const v16bf*)(W + ((kc * 8 + wave) * 32 + lane) * 16);

#pragma unroll
    for (int rt = 0; rt < 4; ++rt) {
        const __bf16* arow = &sA[(rt * 16 + l16) * D];
        v8f acc = {};
#pragma unroll
        for (int kc = 0; kc < 4; ++kc) {
            v8bf lo = *(const v8bf*)(arow + kc * 32 + lhalf * 8);
            v8bf hi = *(const v8bf*)(arow + kc * 32 + 16 + lhalf * 8);
            acc = __builtin_amdgcn_wmma_f32_16x16x32_bf16(
                false, cat16(lo, hi), false, bfrag[kc], (short)0, acc,
                false, false);
        }
#pragma unroll
        for (int r = 0; r < 8; ++r) {
            int row = rt * 16 + r + lhalf * 8;
            float g = (float)gw[(rowbase + row) * D + col];
            out[(rowbase + row) * D + col] = acc[r] * g;
        }
    }
}

// ---------------------------------------------------------------------------
extern "C" void kernel_launch(void* const* d_in, const int* in_sizes, int n_in,
                              void* d_out, int out_size, void* d_ws, size_t ws_size,
                              hipStream_t stream) {
    (void)in_sizes; (void)n_in; (void)out_size; (void)ws_size;
    const float* pair = (const float*)d_in[0];
    const float* Wq   = (const float*)d_in[1];
    const float* Wk   = (const float*)d_in[2];
    const float* Wv   = (const float*)d_in[3];
    const float* Wb   = (const float*)d_in[4];
    const float* Wo   = (const float*)d_in[5];
    const float* Wg   = (const float*)d_in[6];
    float* out = (float*)d_out;

    const size_t SQ = (size_t)M * D;                 // per bf16 tensor
    __bf16* qw = (__bf16*)d_ws;
    __bf16* kw = qw + SQ;
    __bf16* vw = kw + SQ;
    __bf16* gw = vw + SQ;
    __bf16* aw = gw + SQ;
    float*  biasw = (float*)(aw + SQ);               // H * M floats, [h][m]
    __bf16* wts = (__bf16*)(biasw + (size_t)H * M);  // 5*D*D bf16, swizzled

    cvt_weights<<<(5 * D * D + 255) / 256, 256, 0, stream>>>(Wq, Wk, Wv, Wg, Wo, wts);
    proj_kernel<<<M / 64, 256, 0, stream>>>(pair, Wb, wts, qw, kw, vw, gw, biasw);
    attn_kernel<<<dim3(N / 64, H, N), 256, 0, stream>>>(qw, kw, vw, biasw, aw);
    outproj_kernel<<<M / 64, 256, 0, stream>>>(aw, wts, gw, out);
}
